// ImprovedGNN_52879637348874
// MI455X (gfx1250) — matrix operand order repaired and checked by reading, not compiled, not run on hardware
//
#include <hip/hip_runtime.h>
#include <cstdint>

// ---------- types for WMMA ----------
typedef __attribute__((ext_vector_type(8)))  float   v8f;
typedef __attribute__((ext_vector_type(8)))  __bf16  v8bf;
typedef __attribute__((ext_vector_type(16))) __bf16  v16bf;

// ---------- helpers ----------
__device__ __forceinline__ unsigned short f2bf(float f) {
    unsigned int u = __float_as_uint(f);
    unsigned int r = u + 0x7FFFu + ((u >> 16) & 1u);   // round-to-nearest-even
    return (unsigned short)(r >> 16);
}

// ---------- elementwise convert x (f32) -> bf16 ----------
__global__ void gcn_convert_bf16(const float* __restrict__ x,
                                 unsigned short* __restrict__ xb, int n) {
    int t = blockIdx.x * blockDim.x + threadIdx.x;
    if (t < n) xb[t] = f2bf(x[t]);
}

// ---------- pack weight [K,Nout] f32 into WMMA-B fragment order (bf16) ----------
// Tile (kt,nt) -> 512 bf16. Within tile: lane (0..31) * 16 elems.
//   lane<16 : column n = nt*16+lane,       K = kt*32 + (j<8 ? j : 8+j)      (j=0..15)
//   lane>=16: column n = nt*16+(lane-16),  K = kt*32 + 8 + (j<8 ? j : 8+j)
__global__ void gcn_pack_w(const float* __restrict__ W,
                           unsigned short* __restrict__ wp, int K, int Nout) {
    int t = blockIdx.x * blockDim.x + threadIdx.x;
    int total = K * Nout;
    if (t >= total) return;
    int ntiles = Nout >> 4;
    int tile = t >> 9;
    int r    = t & 511;
    int lane = r >> 4;
    int j    = r & 15;
    int kt = tile / ntiles, nt = tile % ntiles;
    int n = (nt << 4) + (lane & 15);
    int k = (kt << 5) + ((lane < 16) ? 0 : 8) + ((j < 8) ? j : (8 + j));
    wp[t] = f2bf(W[(size_t)k * Nout + n]);
}

// ---------- degree / norm ----------
__global__ void gcn_init_deg(float* __restrict__ deg, int n) {
    int t = blockIdx.x * blockDim.x + threadIdx.x;
    if (t < n) deg[t] = 1.0f;                   // +1 self loop
}
__global__ void gcn_count_deg(const int* __restrict__ dst,
                              float* __restrict__ deg, int E) {
    int t = blockIdx.x * blockDim.x + threadIdx.x;
    if (t < E) atomicAdd(&deg[dst[t]], 1.0f);
}
__global__ void gcn_rsqrt(float* __restrict__ deg, int n) {
    int t = blockIdx.x * blockDim.x + threadIdx.x;
    if (t < n) deg[t] = rsqrtf(deg[t]);
}
__global__ void gcn_coef(const int* __restrict__ src, const int* __restrict__ dst,
                         const float* __restrict__ dinv,
                         float* __restrict__ coef, int E) {
    int t = blockIdx.x * blockDim.x + threadIdx.x;
    if (t < E) coef[t] = dinv[src[t]] * dinv[dst[t]];
}

// ---------- bf16 WMMA GEMM: h[M,Nout] = xb[M,K](bf16,ld=lda) @ Wpacked ----------
// One wave computes a 16x64 strip (4 accumulators): one A fragment feeds 4 WMMAs.
__global__ void gcn_gemm_bf16(const unsigned short* __restrict__ xb, int lda,
                              const unsigned short* __restrict__ wp,
                              float* __restrict__ hout, int M, int K, int Nout) {
    int lane = threadIdx.x & 31;
    int wave = threadIdx.x >> 5;
    int mtiles = M >> 4;
    int ntiles = Nout >> 4;
    int mtile  = blockIdx.x * (blockDim.x >> 5) + wave;
    int ntile0 = blockIdx.y << 2;               // 4 consecutive n-tiles per wave
    if (mtile >= mtiles) return;

    int mbase = mtile << 4;
    int row   = mbase + (lane & 15);
    int koff  = (lane >= 16) ? 8 : 0;

    v8f acc0 = {}, acc1 = {}, acc2 = {}, acc3 = {};
    int ktiles = K >> 5;
    for (int kt = 0; kt < ktiles; ++kt) {
        // A fragment: 16-bit A 16x32 layout (two contiguous 16B chunks / lane)
        const unsigned short* ap = xb + (size_t)row * lda + (kt << 5) + koff;
        v8bf alo = *(const v8bf*)(ap);
        v8bf ahi = *(const v8bf*)(ap + 16);
        v16bf a = __builtin_shufflevector(alo, ahi,
                     0,1,2,3,4,5,6,7,8,9,10,11,12,13,14,15);
        // B fragments: pre-packed, one contiguous 32B load / lane each
        const unsigned short* bp =
            wp + ((size_t)(kt * ntiles + ntile0) << 9) + (lane << 4);
        v16bf b0 = *(const v16bf*)(bp);
        v16bf b1 = *(const v16bf*)(bp + 512);
        v16bf b2 = *(const v16bf*)(bp + 1024);
        v16bf b3 = *(const v16bf*)(bp + 1536);
        acc0 = __builtin_amdgcn_wmma_f32_16x16x32_bf16(
                   false, a, false, b0, (short)0, acc0, false, false);
        acc1 = __builtin_amdgcn_wmma_f32_16x16x32_bf16(
                   false, a, false, b1, (short)0, acc1, false, false);
        acc2 = __builtin_amdgcn_wmma_f32_16x16x32_bf16(
                   false, a, false, b2, (short)0, acc2, false, false);
        acc3 = __builtin_amdgcn_wmma_f32_16x16x32_bf16(
                   false, a, false, b3, (short)0, acc3, false, false);
    }
    // C/D layout: lanes 0-15 -> M=v, N=lane ; lanes 16-31 -> M=v+8, N=lane-16
    int rofs = (lane >= 16) ? 8 : 0;
    int col  = (ntile0 << 4) + (lane & 15);
#pragma unroll
    for (int v = 0; v < 8; ++v) {
        float* dp = hout + (size_t)(mbase + v + rofs) * Nout + col;
        dp[0]  = acc0[v];
        dp[16] = acc1[v];
        dp[32] = acc2[v];
        dp[48] = acc3[v];
    }
}

// ---------- edge scatter: agg[dst] += h[src]*coef, one wave per edge ----------
// float4 per lane-iteration: one b128 load + 4 f32 atomics (L2-resident).
__global__ void gcn_scatter(const float* __restrict__ h,
                            float* __restrict__ agg,
                            const int* __restrict__ src,
                            const int* __restrict__ dst,
                            const float* __restrict__ coef, int E, int d) {
    int wid  = (blockIdx.x * blockDim.x + threadIdx.x) >> 5;
    int lane = threadIdx.x & 31;
    if (wid >= E) return;
    int s = src[wid], t = dst[wid];
    float c = coef[wid];
    const float* hp = h   + (size_t)s * d;
    float*       ap = agg + (size_t)t * d;
    for (int f = lane * 4; f < d; f += 128) {
        float4 hv = *(const float4*)(hp + f);
        atomicAdd(&ap[f + 0], hv.x * c);
        atomicAdd(&ap[f + 1], hv.y * c);
        atomicAdd(&ap[f + 2], hv.z * c);
        atomicAdd(&ap[f + 3], hv.w * c);
    }
}

// ---------- finalize mid layer: self-loop + bias, LayerNorm, SiLU -> bf16 ----------
__global__ void gcn_finalize_mid(const float* __restrict__ agg,
                                 const float* __restrict__ h,
                                 const float* __restrict__ dinv,
                                 const float* __restrict__ bias,
                                 const float* __restrict__ gamma,
                                 const float* __restrict__ beta,
                                 unsigned short* __restrict__ xb) {
    const int D = 256;
    int i = blockIdx.x;
    int f = threadIdx.x;
    float di = dinv[i];
    float v = agg[(size_t)i * D + f] + h[(size_t)i * D + f] * di * di + bias[f];

    __shared__ float red[256];
    red[f] = v; __syncthreads();
    for (int s = 128; s > 0; s >>= 1) {
        if (f < s) red[f] += red[f + s];
        __syncthreads();
    }
    float mu = red[0] * (1.0f / 256.0f);
    __syncthreads();
    float dv = v - mu;
    red[f] = dv * dv; __syncthreads();
    for (int s = 128; s > 0; s >>= 1) {
        if (f < s) red[f] += red[f + s];
        __syncthreads();
    }
    float var = red[0] * (1.0f / 256.0f);
    float y = dv * rsqrtf(var + 1e-5f) * gamma[f] + beta[f];
    float o = y / (1.0f + expf(-y));          // SiLU
    xb[(size_t)i * D + f] = f2bf(o);
}

// ---------- finalize output layer (no LN/SiLU), write f32 ----------
__global__ void gcn_finalize_out(const float* __restrict__ agg,
                                 const float* __restrict__ h,
                                 const float* __restrict__ dinv,
                                 const float* __restrict__ bias,
                                 float* __restrict__ out) {
    const int D = 128;
    int i = blockIdx.x;
    int f = threadIdx.x;
    float di = dinv[i];
    out[(size_t)i * D + f] = agg[(size_t)i * D + f]
                           + h[(size_t)i * D + f] * di * di + bias[f];
}

extern "C" void kernel_launch(void* const* d_in, const int* in_sizes, int n_in,
                              void* d_out, int out_size, void* d_ws, size_t ws_size,
                              hipStream_t stream) {
    const float* x     = (const float*)d_in[0];
    const int*   ei    = (const int*)  d_in[1];
    const float* W_in  = (const float*)d_in[2];
    const float* b_in  = (const float*)d_in[3];
    const float* W_mid = (const float*)d_in[4];
    const float* b_mid = (const float*)d_in[5];
    const float* W_out = (const float*)d_in[6];
    const float* b_out = (const float*)d_in[7];
    const float* gamma = (const float*)d_in[8];
    const float* beta  = (const float*)d_in[9];
    float* out = (float*)d_out;

    const int N = in_sizes[0] / 128;
    const int E = in_sizes[1] / 2;
    const int* src = ei;
    const int* dst = ei + E;

    // ---- workspace carve-out ----
    char* ws = (char*)d_ws;
    size_t off = 0;
    auto carve = [&](size_t bytes) -> char* {
        char* p = ws + off;
        off = (off + bytes + 255) & ~(size_t)255;
        return p;
    };
    unsigned short* xb  = (unsigned short*)carve((size_t)N * 256 * 2);
    float*          h   = (float*)         carve((size_t)N * 256 * 4);
    float*          agg = (float*)         carve((size_t)N * 256 * 4);
    float*          din = (float*)         carve((size_t)N * 4);
    float*          cof = (float*)         carve((size_t)E * 4);
    unsigned short* wp  = (unsigned short*)carve((size_t)(32768 + 8 * 65536 + 32768) * 2);
    unsigned short* wp_in  = wp;
    unsigned short* wp_mid = wp + 32768;
    unsigned short* wp_out = wp + 32768 + 8 * 65536;

    const int T = 256;

    // 1) convert input activations to bf16 (ld = 128)
    gcn_convert_bf16<<<(N * 128 + T - 1) / T, T, 0, stream>>>(x, xb, N * 128);

    // 2) pack all weights into WMMA-B fragment order (once per replay)
    gcn_pack_w<<<(128 * 256 + T - 1) / T, T, 0, stream>>>(W_in, wp_in, 128, 256);
    for (int i = 0; i < 8; ++i)
        gcn_pack_w<<<(256 * 256 + T - 1) / T, T, 0, stream>>>(
            W_mid + (size_t)i * 65536, wp_mid + (size_t)i * 65536, 256, 256);
    gcn_pack_w<<<(256 * 128 + T - 1) / T, T, 0, stream>>>(W_out, wp_out, 256, 128);

    // 3) degrees -> dinv -> per-edge coefficients (shared by all layers)
    gcn_init_deg <<<(N + T - 1) / T, T, 0, stream>>>(din, N);
    gcn_count_deg<<<(E + T - 1) / T, T, 0, stream>>>(dst, din, E);
    gcn_rsqrt    <<<(N + T - 1) / T, T, 0, stream>>>(din, N);
    gcn_coef     <<<(E + T - 1) / T, T, 0, stream>>>(src, dst, din, cof, E);

    const int mtiles = N / 16;
    dim3 blkG(256);
    // 8 waves per block; each wave: one mtile x 4 ntiles
    auto run_layer = [&](int lda, const unsigned short* wlay, int K, int Nout,
                         const float* bias, const float* g, const float* b,
                         bool last) {
        dim3 grdG((mtiles + 7) / 8, (Nout / 16) / 4);
        gcn_gemm_bf16<<<grdG, blkG, 0, stream>>>(xb, lda, wlay, h, N, K, Nout);
        hipMemsetAsync(agg, 0, (size_t)N * Nout * 4, stream);
        gcn_scatter<<<(E + 7) / 8, 256, 0, stream>>>(h, agg, src, dst, cof, E, Nout);
        if (!last) {
            gcn_finalize_mid<<<N, 256, 0, stream>>>(agg, h, din, bias, g, b, xb);
        } else {
            gcn_finalize_out<<<N, 128, 0, stream>>>(agg, h, din, bias, out);
        }
    };

    // layer 0: 128 -> 256
    run_layer(128, wp_in, 128, 256, b_in, gamma + 0, beta + 0, false);
    // mid layers: 256 -> 256
    for (int i = 0; i < 8; ++i)
        run_layer(256, wp_mid + (size_t)i * 65536, 256, 256,
                  b_mid + (size_t)i * 256,
                  gamma + (size_t)(i + 1) * 256,
                  beta  + (size_t)(i + 1) * 256, false);
    // output layer: 256 -> 128
    run_layer(256, wp_out, 256, 128, b_out, nullptr, nullptr, true);
}
